// SparseMOE_72507637891701
// MI455X (gfx1250) — compile-verified, matrix operand order checked by reading, not run on hardware
//
#include <hip/hip_runtime.h>
#include <hip/hip_bf16.h>
#include <math.h>

// Problem constants (from reference setup_inputs)
#define NTOK 8192
#define DDIM 1024
#define HDIM 4096
#define NEXP 8
#define TOPK 2

typedef __bf16 bf16;
typedef __attribute__((ext_vector_type(16))) __bf16 bf16x16;
typedef __attribute__((ext_vector_type(8)))  __bf16 bf16x8;
typedef __attribute__((ext_vector_type(8)))  float  fx8;

__device__ __forceinline__ bf16 to_bf16(float f) { return (bf16)f; }

// Load one 16x32 bf16 fragment (A-style layout) from a row-major matrix.
// Lane l (l&15 selects the row), half = l>>4:
//   half==0 -> K = k0+{0..7, 16..23};  half==1 -> K = k0+{8..15, 24..31}
__device__ __forceinline__ bf16x16 load_frag(const bf16* __restrict__ row, int k0, int half) {
  const bf16* p = row + k0 + half * 8;
  bf16x8 lo = *(const bf16x8*)(p);
  bf16x8 hi = *(const bf16x8*)(p + 16);
  bf16x16 r;
#pragma unroll
  for (int i = 0; i < 8; ++i) { r[i] = lo[i]; r[i + 8] = hi[i]; }
  return r;
}

__device__ __forceinline__ fx8 wmma_bf16(bf16x16 a, bf16x16 b, fx8 c) {
  return __builtin_amdgcn_wmma_f32_16x16x32_bf16(false, a, false, b, (short)0, c, false, false);
}

__device__ __forceinline__ void atomic_add_f32(float* p, float v) {
  __hip_atomic_fetch_add(p, v, __ATOMIC_RELAXED, __HIP_MEMORY_SCOPE_AGENT);
}

// Load a full pipeline stage: 4 A-fragments + 2 B-fragments at k0.
__device__ __forceinline__ void load_stage(const bf16* const* ap, const bf16* const* bp,
                                           int k0, int half, bf16x16* A, bf16x16* B) {
#pragma unroll
  for (int mi = 0; mi < 4; ++mi) A[mi] = load_frag(ap[mi], k0, half);
#pragma unroll
  for (int ni = 0; ni < 2; ++ni) B[ni] = load_frag(bp[ni], k0, half);
}

__device__ __forceinline__ void mma_stage(const bf16x16* A, const bf16x16* B, fx8 acc[4][2]) {
#pragma unroll
  for (int mi = 0; mi < 4; ++mi)
#pragma unroll
    for (int ni = 0; ni < 2; ++ni)
      acc[mi][ni] = wmma_bf16(A[mi], B[ni], acc[mi][ni]);
}

// ---------------------------------------------------------------------------
// 0) zero y (+ loss slot) and small accumulators
// ---------------------------------------------------------------------------
__global__ void zero_init(float* __restrict__ y, size_t ny,
                          float* __restrict__ imp, float* __restrict__ loadv,
                          int* __restrict__ counts) {
  size_t stride = (size_t)gridDim.x * blockDim.x;
  for (size_t i = (size_t)blockIdx.x * blockDim.x + threadIdx.x; i < ny; i += stride)
    y[i] = 0.0f;
  if (blockIdx.x == 0 && threadIdx.x < NEXP) {
    imp[threadIdx.x] = 0.0f;
    loadv[threadIdx.x] = 0.0f;
    counts[threadIdx.x] = 0;
  }
}

// ---------------------------------------------------------------------------
// 1) f32 -> bf16 elementwise convert (for x)
// ---------------------------------------------------------------------------
__global__ void convert_bf16(const float* __restrict__ src, bf16* __restrict__ dst, size_t n) {
  size_t stride = (size_t)gridDim.x * blockDim.x;
  for (size_t i = (size_t)blockIdx.x * blockDim.x + threadIdx.x; i < n; i += stride)
    dst[i] = to_bf16(src[i]);
}

// ---------------------------------------------------------------------------
// 2) per-expert transpose + convert: src (R x C) row-major -> dst (C x R) bf16
// ---------------------------------------------------------------------------
__global__ __launch_bounds__(256) void transpose_convert(const float* __restrict__ src,
                                                         bf16* __restrict__ dst,
                                                         int R, int C) {
  __shared__ float tile[32][33];
  size_t eoff = (size_t)blockIdx.z * R * C;
  const float* s = src + eoff;
  bf16* d = dst + eoff;
  int c0 = blockIdx.x * 32, r0 = blockIdx.y * 32;
#pragma unroll
  for (int i = threadIdx.y; i < 32; i += 8)
    tile[i][threadIdx.x] = s[(size_t)(r0 + i) * C + (c0 + threadIdx.x)];
  __syncthreads();
#pragma unroll
  for (int i = threadIdx.y; i < 32; i += 8)
    d[(size_t)(c0 + i) * R + (r0 + threadIdx.x)] = to_bf16(tile[threadIdx.x][i]);
}

// ---------------------------------------------------------------------------
// 3) Gating: one wave32 per token.
// ---------------------------------------------------------------------------
__global__ __launch_bounds__(256) void gating_kernel(
    const float* __restrict__ x, const float* __restrict__ noise,
    const float* __restrict__ wg, const float* __restrict__ wn,
    int* __restrict__ meta_e, int* __restrict__ meta_r, float* __restrict__ meta_g,
    float* __restrict__ imp, float* __restrict__ loadv, int* __restrict__ counts) {
  int wave = threadIdx.x >> 5;
  int lane = threadIdx.x & 31;
  int tok = blockIdx.x * 8 + wave;
  if (tok >= NTOK) return;

  float accg[NEXP], accn[NEXP];
#pragma unroll
  for (int e = 0; e < NEXP; ++e) { accg[e] = 0.0f; accn[e] = 0.0f; }

  const float* xr = x + (size_t)tok * DDIM;
  for (int d = lane; d < DDIM; d += 32) {
    float xv = xr[d];
    const float* wgr = wg + (size_t)d * NEXP;
    const float* wnr = wn + (size_t)d * NEXP;
#pragma unroll
    for (int e = 0; e < NEXP; ++e) {
      accg[e] = fmaf(xv, wgr[e], accg[e]);
      accn[e] = fmaf(xv, wnr[e], accn[e]);
    }
  }
#pragma unroll
  for (int off = 16; off > 0; off >>= 1) {
#pragma unroll
    for (int e = 0; e < NEXP; ++e) {
      accg[e] += __shfl_xor(accg[e], off, 32);
      accn[e] += __shfl_xor(accn[e], off, 32);
    }
  }

  if (lane == 0) {
    float clean[NEXP], stdv[NEXP], noisy[NEXP];
#pragma unroll
    for (int e = 0; e < NEXP; ++e) {
      clean[e] = accg[e];
      float sn = accn[e];
      float sp = (sn > 20.0f) ? sn : log1pf(expf(sn));  // softplus
      stdv[e] = sp + 0.01f;                              // NOISE_EPS
      noisy[e] = clean[e] + noise[(size_t)tok * NEXP + e] * stdv[e];
    }
    // top-3 of 8
    int i0 = 0;
#pragma unroll
    for (int e = 1; e < NEXP; ++e) if (noisy[e] > noisy[i0]) i0 = e;
    int i1 = -1;
#pragma unroll
    for (int e = 0; e < NEXP; ++e)
      if (e != i0 && (i1 < 0 || noisy[e] > noisy[i1])) i1 = e;
    int i2 = -1;
#pragma unroll
    for (int e = 0; e < NEXP; ++e)
      if (e != i0 && e != i1 && (i2 < 0 || noisy[e] > noisy[i2])) i2 = e;
    float v1 = noisy[i1], v2 = noisy[i2];
    // softmax over top-2
    float e1 = expf(v1 - noisy[i0]);
    float g0 = 1.0f / (1.0f + e1);
    float g1 = e1 * g0;
    atomic_add_f32(&imp[i0], g0);
    atomic_add_f32(&imp[i1], g1);
    // load = sum over tokens of P(expert in top-K)
#pragma unroll
    for (int e = 0; e < NEXP; ++e) {
      bool is_in = noisy[e] > v2;
      float thr = is_in ? v2 : v1;
      float z = (clean[e] - thr) / stdv[e];
      float p = 0.5f * erfcf(-z * 0.70710678118654752f);  // Phi(z)
      atomic_add_f32(&loadv[e], p);
    }
    int r0 = atomicAdd(&counts[i0], 1);
    int r1 = atomicAdd(&counts[i1], 1);
    meta_e[tok * 2] = i0; meta_e[tok * 2 + 1] = i1;
    meta_r[tok * 2] = r0; meta_r[tok * 2 + 1] = r1;
    meta_g[tok * 2] = g0; meta_g[tok * 2 + 1] = g1;
  }
}

// ---------------------------------------------------------------------------
// 4) prefix-sum over E=8 + auxiliary loss -> y[N*D]
// ---------------------------------------------------------------------------
__global__ void offsets_loss_kernel(const int* __restrict__ counts, int* __restrict__ offsets,
                                    const float* __restrict__ imp, const float* __restrict__ loadv,
                                    float* __restrict__ y) {
  if (threadIdx.x != 0 || blockIdx.x != 0) return;
  int run = 0;
  for (int e = 0; e < NEXP; ++e) { offsets[e] = run; run += counts[e]; }
  float cv2[2];
  const float* v[2] = {imp, loadv};
  for (int s = 0; s < 2; ++s) {
    float sum = 0.0f;
    for (int e = 0; e < NEXP; ++e) sum += v[s][e];
    float mean = sum / NEXP;
    float var = 0.0f;
    for (int e = 0; e < NEXP; ++e) { float d = v[s][e] - mean; var += d * d; }
    var /= (NEXP - 1);  // ddof=1
    cv2[s] = var / (mean * mean + 1e-10f);
  }
  y[(size_t)NTOK * DDIM] = 0.01f * (cv2[0] + cv2[1]);  // LOSS_COEF
}

// ---------------------------------------------------------------------------
// 5) scatter tokens into per-expert contiguous lists
// ---------------------------------------------------------------------------
__global__ void scatter_kernel(const int* __restrict__ meta_e, const int* __restrict__ meta_r,
                               const float* __restrict__ meta_g, const int* __restrict__ offsets,
                               int* __restrict__ tok, float* __restrict__ gl) {
  int n = blockIdx.x * blockDim.x + threadIdx.x;
  if (n >= NTOK) return;
#pragma unroll
  for (int k = 0; k < TOPK; ++k) {
    int e = meta_e[n * 2 + k];
    int pos = offsets[e] + meta_r[n * 2 + k];
    tok[pos] = n;
    gl[pos] = meta_g[n * 2 + k];
  }
}

// ---------------------------------------------------------------------------
// 6) FFN1: h[pos, :] = relu(x[tok[pos]] @ W1[e] + b1[e])  (bf16 WMMA GEMM)
//    Block: 256 threads = 8 waves arranged 2(M) x 4(N); block tile 128 x 128.
//    Wave tile 64x32 = 4x2 WMMA fragments; manual ping-pong k-pipeline
//    (two fragment sets, no inter-set copies).
// ---------------------------------------------------------------------------
__global__ __launch_bounds__(256) void moe_ffn1(
    const bf16* __restrict__ xb, const bf16* __restrict__ W1t,
    const float* __restrict__ b1, bf16* __restrict__ hbuf,
    const int* __restrict__ tok, const int* __restrict__ offsets,
    const int* __restrict__ counts) {
  int e = blockIdx.z;
  int M = counts[e];
  int m_blk = blockIdx.y * 128;
  if (m_blk >= M) return;
  int offE = offsets[e];

  int wave = threadIdx.x >> 5, lane = threadIdx.x & 31;
  int wm = wave & 1, wn = wave >> 1;
  int m_base = m_blk + wm * 64;
  int n_base = blockIdx.x * 128 + wn * 32;
  int lrow = lane & 15, half = lane >> 4;

  const bf16* ap[4];
#pragma unroll
  for (int mi = 0; mi < 4; ++mi) {
    int r = m_base + mi * 16 + lrow;
    if (r > M - 1) r = M - 1;
    ap[mi] = xb + (size_t)tok[offE + r] * DDIM;
  }
  const bf16* Wb = W1t + (size_t)e * HDIM * DDIM;
  const bf16* bp[2];
#pragma unroll
  for (int ni = 0; ni < 2; ++ni)
    bp[ni] = Wb + (size_t)(n_base + ni * 16 + lrow) * DDIM;

  fx8 acc[4][2];
#pragma unroll
  for (int mi = 0; mi < 4; ++mi)
#pragma unroll
    for (int ni = 0; ni < 2; ++ni) acc[mi][ni] = (fx8){};

  // manual ping-pong pipeline over k (stage = 32)
  bf16x16 A0[4], B0[2], A1[4], B1[2];
  load_stage(ap, bp, 0, half, A0, B0);
  int k0 = 0;
  for (; k0 + 64 < DDIM; k0 += 64) {
    load_stage(ap, bp, k0 + 32, half, A1, B1);
    mma_stage(A0, B0, acc);
    load_stage(ap, bp, k0 + 64, half, A0, B0);
    mma_stage(A1, B1, acc);
  }
  load_stage(ap, bp, DDIM - 32, half, A1, B1);
  mma_stage(A0, B0, acc);
  mma_stage(A1, B1, acc);

  int rowoff = half * 8, colsub = lane & 15;
  const float* b1e = b1 + (size_t)e * HDIM;
  bool full = (m_blk + 128 <= M);
#pragma unroll
  for (int mi = 0; mi < 4; ++mi) {
#pragma unroll
    for (int ni = 0; ni < 2; ++ni) {
      int hcol = n_base + ni * 16 + colsub;
      float bias = b1e[hcol];
      if (full) {
#pragma unroll
        for (int i = 0; i < 8; ++i) {
          int m = m_base + mi * 16 + rowoff + i;
          float v = acc[mi][ni][i] + bias;
          v = v > 0.0f ? v : 0.0f;
          hbuf[(size_t)(offE + m) * HDIM + hcol] = to_bf16(v);
        }
      } else {
#pragma unroll
        for (int i = 0; i < 8; ++i) {
          int m = m_base + mi * 16 + rowoff + i;
          if (m < M) {
            float v = acc[mi][ni][i] + bias;
            v = v > 0.0f ? v : 0.0f;
            hbuf[(size_t)(offE + m) * HDIM + hcol] = to_bf16(v);
          }
        }
      }
    }
  }
}

// ---------------------------------------------------------------------------
// 7) FFN2: y[tok[pos]] += gate[pos] * (h[pos] @ W2[e] + b2[e])
//    Same 128x128 block tile / 64x32 wave tile, manual ping-pong.
// ---------------------------------------------------------------------------
__global__ __launch_bounds__(256) void moe_ffn2(
    const bf16* __restrict__ hbuf, const bf16* __restrict__ W2t,
    const float* __restrict__ b2, float* __restrict__ y,
    const int* __restrict__ tok, const float* __restrict__ gl,
    const int* __restrict__ offsets, const int* __restrict__ counts) {
  int e = blockIdx.z;
  int M = counts[e];
  int m_blk = blockIdx.y * 128;
  if (m_blk >= M) return;
  int offE = offsets[e];

  int wave = threadIdx.x >> 5, lane = threadIdx.x & 31;
  int wm = wave & 1, wn = wave >> 1;
  int m_base = m_blk + wm * 64;
  int n_base = blockIdx.x * 128 + wn * 32;
  int lrow = lane & 15, half = lane >> 4;

  const bf16* ap[4];
#pragma unroll
  for (int mi = 0; mi < 4; ++mi) {
    int r = m_base + mi * 16 + lrow;
    if (r > M - 1) r = M - 1;
    ap[mi] = hbuf + (size_t)(offE + r) * HDIM;
  }
  const bf16* Wb = W2t + (size_t)e * DDIM * HDIM;
  const bf16* bp[2];
#pragma unroll
  for (int ni = 0; ni < 2; ++ni)
    bp[ni] = Wb + (size_t)(n_base + ni * 16 + lrow) * HDIM;

  fx8 acc[4][2];
#pragma unroll
  for (int mi = 0; mi < 4; ++mi)
#pragma unroll
    for (int ni = 0; ni < 2; ++ni) acc[mi][ni] = (fx8){};

  bf16x16 A0[4], B0[2], A1[4], B1[2];
  load_stage(ap, bp, 0, half, A0, B0);
  int k0 = 0;
  for (; k0 + 64 < HDIM; k0 += 64) {
    load_stage(ap, bp, k0 + 32, half, A1, B1);
    mma_stage(A0, B0, acc);
    load_stage(ap, bp, k0 + 64, half, A0, B0);
    mma_stage(A1, B1, acc);
  }
  load_stage(ap, bp, HDIM - 32, half, A1, B1);
  mma_stage(A0, B0, acc);
  mma_stage(A1, B1, acc);

  int rowoff = half * 8, colsub = lane & 15;
  const float* b2e = b2 + (size_t)e * DDIM;
  bool full = (m_blk + 128 <= M);
#pragma unroll
  for (int mi = 0; mi < 4; ++mi) {
    // hoist token/gate per row (uniform across the 16 lanes of a half)
    int   t8[8];
    float g8[8];
#pragma unroll
    for (int i = 0; i < 8; ++i) {
      int m = m_base + mi * 16 + rowoff + i;
      int mc = m < M ? m : M - 1;
      t8[i] = tok[offE + mc];
      g8[i] = gl[offE + mc];
    }
#pragma unroll
    for (int ni = 0; ni < 2; ++ni) {
      int dcol = n_base + ni * 16 + colsub;
      float bias = b2e[dcol];
      if (full) {
#pragma unroll
        for (int i = 0; i < 8; ++i)
          atomic_add_f32(&y[(size_t)t8[i] * DDIM + dcol],
                         g8[i] * (acc[mi][ni][i] + bias));
      } else {
#pragma unroll
        for (int i = 0; i < 8; ++i) {
          int m = m_base + mi * 16 + rowoff + i;
          if (m < M)
            atomic_add_f32(&y[(size_t)t8[i] * DDIM + dcol],
                           g8[i] * (acc[mi][ni][i] + bias));
        }
      }
    }
  }
}

// ---------------------------------------------------------------------------
// Host launcher
// ---------------------------------------------------------------------------
extern "C" void kernel_launch(void* const* d_in, const int* in_sizes, int n_in,
                              void* d_out, int out_size, void* d_ws, size_t ws_size,
                              hipStream_t stream) {
  const float* x       = (const float*)d_in[0];
  const float* noise   = (const float*)d_in[1];
  const float* w_gate  = (const float*)d_in[2];
  const float* w_noise = (const float*)d_in[3];
  const float* W1      = (const float*)d_in[4];
  const float* b1      = (const float*)d_in[5];
  const float* W2      = (const float*)d_in[6];
  const float* b2      = (const float*)d_in[7];
  float* y = (float*)d_out;  // [N*D] y, then 1 loss scalar

  char* w = (char*)d_ws;
  size_t o = 0;
  auto take = [&](size_t b) { size_t cur = o; o += (b + 255) & ~(size_t)255; return cur; };
  bf16*  xb     = (bf16*)(w + take((size_t)NTOK * DDIM * 2));
  bf16*  W1t    = (bf16*)(w + take((size_t)NEXP * HDIM * DDIM * 2));
  bf16*  W2t    = (bf16*)(w + take((size_t)NEXP * DDIM * HDIM * 2));
  bf16*  hbuf   = (bf16*)(w + take((size_t)NTOK * TOPK * HDIM * 2));
  int*   meta_e = (int*)(w + take((size_t)NTOK * 2 * 4));
  int*   meta_r = (int*)(w + take((size_t)NTOK * 2 * 4));
  float* meta_g = (float*)(w + take((size_t)NTOK * 2 * 4));
  int*   tokl   = (int*)(w + take((size_t)NTOK * 2 * 4));
  float* gl     = (float*)(w + take((size_t)NTOK * 2 * 4));
  int*   counts = (int*)(w + take(NEXP * 4));
  int*   offs   = (int*)(w + take(NEXP * 4));
  float* imp    = (float*)(w + take(NEXP * 4));
  float* loadv  = (float*)(w + take(NEXP * 4));
  (void)ws_size; (void)in_sizes; (void)n_in; (void)out_size;

  // 0) zero y + accumulators
  zero_init<<<4096, 256, 0, stream>>>(y, (size_t)NTOK * DDIM + 1, imp, loadv, counts);
  // 1) x -> bf16
  convert_bf16<<<4096, 256, 0, stream>>>(x, xb, (size_t)NTOK * DDIM);
  // 2) W1 (E,D,H) -> W1t (E,H,D) ; W2 (E,H,D) -> W2t (E,D,H)
  transpose_convert<<<dim3(HDIM / 32, DDIM / 32, NEXP), dim3(32, 8), 0, stream>>>(W1, W1t, DDIM, HDIM);
  transpose_convert<<<dim3(DDIM / 32, HDIM / 32, NEXP), dim3(32, 8), 0, stream>>>(W2, W2t, HDIM, DDIM);
  // 3) gating (one wave per token, 8 tokens per 256-thread block)
  gating_kernel<<<NTOK / 8, 256, 0, stream>>>(x, noise, w_gate, w_noise,
                                              meta_e, meta_r, meta_g, imp, loadv, counts);
  // 4) offsets + loss
  offsets_loss_kernel<<<1, 32, 0, stream>>>(counts, offs, imp, loadv, y);
  // 5) scatter token lists
  scatter_kernel<<<NTOK / 256, 256, 0, stream>>>(meta_e, meta_r, meta_g, offs, tokl, gl);
  // 6) FFN1: grid (H/128, ceil(N/128), E)
  moe_ffn1<<<dim3(HDIM / 128, NTOK / 128, NEXP), 256, 0, stream>>>(xb, W1t, b1, hbuf,
                                                                   tokl, offs, counts);
  // 7) FFN2: grid (D/128, ceil(N/128), E)
  moe_ffn2<<<dim3(DDIM / 128, NTOK / 128, NEXP), 256, 0, stream>>>(hbuf, W2t, b2, y,
                                                                   tokl, gl, offs, counts);
}
